// PatchcoreModel_43164421325483
// MI455X (gfx1250) — matrix operand order repaired and checked by reading, not compiled, not run on hardware
//
#include <hip/hip_runtime.h>
#include <hip/hip_bf16.h>
#include <math.h>

typedef __attribute__((ext_vector_type(16))) _Float16 v16h;
typedef __attribute__((ext_vector_type(8)))  _Float16 h8;
typedef __attribute__((ext_vector_type(8)))  float    v8f;

#define D_DIM   384
#define N_ROWS  16384
#define M_ROWS  16384
#define BATCH   4
#define GRID_WH 64
#define OUT_SZ  224
#define NSPLIT  8
#define COLS_PER_SPLIT (M_ROWS / NSPLIT)   // 2048
#define COL_TILES      (COLS_PER_SPLIT/16) // 128
#define KTILES         (D_DIM / 32)        // 12

// ---------------------------------------------------------------------------
// fp32 -> fp16 hi/lo split (x = h + l, near-fp32 via 3 f16 WMMA products)
// ---------------------------------------------------------------------------
__global__ __launch_bounds__(256) void split_kernel(const float* __restrict__ src,
                                                    _Float16* __restrict__ hi,
                                                    _Float16* __restrict__ lo, int n) {
    int i = blockIdx.x * 256 + threadIdx.x;
    if (i >= n) return;
    float x = src[i];
    _Float16 h = (_Float16)x;
    hi[i] = h;
    lo[i] = (_Float16)(x - (float)h);
}

// row-wise squared norms, one wave per row
__global__ __launch_bounds__(256) void row_norm_kernel(const float* __restrict__ X,
                                                       float* __restrict__ out, int rows) {
    int wv   = (blockIdx.x * 256 + threadIdx.x) >> 5;
    int lane = threadIdx.x & 31;
    if (wv >= rows) return;
    const float* r = X + (size_t)wv * D_DIM;
    float s = 0.f;
    for (int k = lane; k < D_DIM; k += 32) { float v = r[k]; s += v * v; }
#pragma unroll
    for (int m = 16; m >= 1; m >>= 1) s += __shfl_xor(s, m, 32);
    if (lane == 0) out[wv] = s;
}

// ---------------------------------------------------------------------------
// Fused distance GEMM + min/argmin.  D = A x B^T via v_wmma_f32_16x16x32_f16,
// A-hi fragments register-resident, B streamed (fits in L2).
// Fragment layouts per CDNA5 ISA 7.12.2 (wave32).
// ---------------------------------------------------------------------------
__device__ __forceinline__ v16h load_fragA(const _Float16* __restrict__ base) {
    // base points at (row, k0 + half*8); elems 0-7 = K+0..7, elems 8-15 = K+16..23
    h8 x = *(const h8*)(base);
    h8 y = *(const h8*)(base + 16);
    v16h a;
#pragma unroll
    for (int i = 0; i < 8; ++i) { a[i] = x[i]; a[i + 8] = y[i]; }
    return a;
}
__device__ __forceinline__ v16h load_fragB(const _Float16* __restrict__ base) {
    // base points at (col, k0 + half*8); elems 0-7 = K+0..7, elems 8-15 = K+16..23
    h8 x = *(const h8*)(base);
    h8 y = *(const h8*)(base + 16);
    v16h b;
#pragma unroll
    for (int i = 0; i < 8; ++i) { b[i] = x[i]; b[i + 8] = y[i]; }
    return b;
}

__global__ __launch_bounds__(256) void dist_min_kernel(
    const _Float16* __restrict__ Eh, const _Float16* __restrict__ El,
    const _Float16* __restrict__ Mh, const _Float16* __restrict__ Ml,
    const float* __restrict__ nE, const float* __restrict__ nM,
    float* __restrict__ pmin, int* __restrict__ pidx)
{
    const int lane = threadIdx.x & 31;
    const int wave = threadIdx.x >> 5;
    const int rb   = blockIdx.x / NSPLIT;
    const int sp   = blockIdx.x % NSPLIT;
    const int rowBase = (rb * 8 + wave) * 16;
    const int colBase = sp * COLS_PER_SPLIT;
    const int hh  = lane >> 4;   // half-wave select
    const int l15 = lane & 15;

    // resident A-hi fragments: 12 x v16h = 96 VGPRs
    v16h ah[KTILES];
    const _Float16* aHrow = Eh + (size_t)(rowBase + l15) * D_DIM + hh * 8;
    const _Float16* aLrow = El + (size_t)(rowBase + l15) * D_DIM + hh * 8;
#pragma unroll
    for (int kk = 0; kk < KTILES; ++kk) ah[kk] = load_fragA(aHrow + kk * 32);

    float rn[8];
#pragma unroll
    for (int j = 0; j < 8; ++j) rn[j] = nE[rowBase + hh * 8 + j];

    float bestv[8]; int besti[8];
#pragma unroll
    for (int j = 0; j < 8; ++j) { bestv[j] = 3.4e38f; besti[j] = 0; }

    for (int ct = 0; ct < COL_TILES; ++ct) {
        const int c0 = colBase + ct * 16;
        const _Float16* bHrow = Mh + (size_t)(c0 + l15) * D_DIM + hh * 8;
        const _Float16* bLrow = Ml + (size_t)(c0 + l15) * D_DIM + hh * 8;
        v8f acc = {};
#pragma unroll
        for (int kk = 0; kk < KTILES; ++kk) {
            v16h bh = load_fragB(bHrow + kk * 32);
            v16h bl = load_fragB(bLrow + kk * 32);
            v16h al = load_fragA(aLrow + kk * 32);
            acc = __builtin_amdgcn_wmma_f32_16x16x32_f16(false, ah[kk], false, bh,
                                                         (short)0, acc, false, false);
            acc = __builtin_amdgcn_wmma_f32_16x16x32_f16(false, al,     false, bh,
                                                         (short)0, acc, false, false);
            acc = __builtin_amdgcn_wmma_f32_16x16x32_f16(false, ah[kk], false, bl,
                                                         (short)0, acc, false, false);
        }
        const float cn = nM[c0 + l15];
        const int   ci = c0 + l15;
#pragma unroll
        for (int j = 0; j < 8; ++j) {
            float d = fmaxf(rn[j] + cn - 2.0f * acc[j], 1e-30f);
            if (d < bestv[j]) { bestv[j] = d; besti[j] = ci; }
        }
    }

    // reduce over the 16 column-lanes of each half-wave group
#pragma unroll
    for (int j = 0; j < 8; ++j) {
        float v = bestv[j]; int i = besti[j];
#pragma unroll
        for (int m = 8; m >= 1; m >>= 1) {
            float ov = __shfl_xor(v, m, 32);
            int   oi = __shfl_xor(i, m, 32);
            if (ov < v || (ov == v && oi < i)) { v = ov; i = oi; }
        }
        if (l15 == 0) {
            int row = rowBase + hh * 8 + j;
            pmin[(size_t)row * NSPLIT + sp] = v;
            pidx[(size_t)row * NSPLIT + sp] = i;
        }
    }
}

__global__ __launch_bounds__(256) void reduce_min_kernel(const float* __restrict__ pmin,
                                                         const int* __restrict__ pidx,
                                                         float* __restrict__ ps,
                                                         int* __restrict__ loc) {
    int row = blockIdx.x * 256 + threadIdx.x;
    if (row >= N_ROWS) return;
    float best = 3.4e38f; int bi = 0x7fffffff;
#pragma unroll
    for (int s = 0; s < NSPLIT; ++s) {
        float v = pmin[(size_t)row * NSPLIT + s];
        int   i = pidx[(size_t)row * NSPLIT + s];
        if (v < best || (v == best && i < bi)) { best = v; bi = i; }
    }
    ps[row]  = sqrtf(best);
    loc[row] = bi;
}

// per-batch argmax of patch scores
__global__ __launch_bounds__(256) void argmax_kernel(const float* __restrict__ ps,
                                                     const int* __restrict__ loc,
                                                     float* __restrict__ score,
                                                     int* __restrict__ maxidx,
                                                     int* __restrict__ nnidx) {
    __shared__ float sv[256]; __shared__ int si[256];
    int b = blockIdx.x, t = threadIdx.x;
    float best = -3.4e38f; int bi = 0;
#pragma unroll
    for (int i = 0; i < 16; ++i) {
        int idx = t * 16 + i;
        float v = ps[b * 4096 + idx];
        if (v > best || (v == best && idx < bi)) { best = v; bi = idx; }
    }
    sv[t] = best; si[t] = bi; __syncthreads();
    for (int s = 128; s >= 1; s >>= 1) {
        if (t < s) {
            if (sv[t + s] > sv[t] || (sv[t + s] == sv[t] && si[t + s] < si[t])) {
                sv[t] = sv[t + s]; si[t] = si[t + s];
            }
        }
        __syncthreads();
    }
    if (t == 0) { score[b] = sv[0]; maxidx[b] = si[0]; nnidx[b] = loc[b * 4096 + si[0]]; }
}

// one row of distances nn_sample vs memory_bank (tiny: 4x16384x384)
__global__ __launch_bounds__(256) void d2_kernel(const float* __restrict__ Mb,
                                                 const float* __restrict__ nM,
                                                 const int* __restrict__ nnidx,
                                                 float* __restrict__ d2) {
    int b = blockIdx.y;
    int m = blockIdx.x * 256 + threadIdx.x;
    if (m >= M_ROWS) return;
    int nn = nnidx[b];
    const float* a = Mb + (size_t)nn * D_DIM;
    const float* c = Mb + (size_t)m * D_DIM;
    float dot = 0.f;
    for (int k = 0; k < D_DIM; ++k) dot += a[k] * c[k];
    d2[(size_t)b * M_ROWS + m] = sqrtf(fmaxf(nM[nn] + nM[m] - 2.0f * dot, 1e-30f));
}

// top-9 smallest + reweighting softmax -> anomaly score (one wave per batch)
__global__ void topk_softmax_kernel(const float* __restrict__ d2,
                                    const float* __restrict__ E,
                                    const float* __restrict__ Mb,
                                    const float* __restrict__ nE,
                                    const float* __restrict__ nM,
                                    const float* __restrict__ score,
                                    const int* __restrict__ maxidx,
                                    float* __restrict__ out_score) {
    __shared__ int chosen[9]; __shared__ float d3[9];
    int b = blockIdx.x, lane = threadIdx.x;
    for (int s = 0; s < 9; ++s) {
        float best = 3.4e38f; int bi = 0x7fffffff;
        for (int m = lane; m < M_ROWS; m += 32) {
            bool skip = false;
            for (int t = 0; t < s; ++t) if (chosen[t] == m) skip = true;
            float v = d2[(size_t)b * M_ROWS + m];
            if (!skip && (v < best || (v == best && m < bi))) { best = v; bi = m; }
        }
#pragma unroll
        for (int mm = 16; mm >= 1; mm >>= 1) {
            float ov = __shfl_xor(best, mm, 32);
            int   oi = __shfl_xor(bi, mm, 32);
            if (ov < best || (ov == best && oi < bi)) { best = ov; bi = oi; }
        }
        if (lane == 0) chosen[s] = bi;
        __syncthreads();
    }
    int q = maxidx[b];   // faithful: embedding indexed FLAT by per-batch argmax
    for (int s = 0; s < 9; ++s) {
        int mrow = chosen[s];
        float dot = 0.f;
        for (int k = lane; k < D_DIM; k += 32)
            dot += E[(size_t)q * D_DIM + k] * Mb[(size_t)mrow * D_DIM + k];
#pragma unroll
        for (int mm = 16; mm >= 1; mm >>= 1) dot += __shfl_xor(dot, mm, 32);
        if (lane == 0) d3[s] = sqrtf(fmaxf(nE[q] + nM[mrow] - 2.0f * dot, 1e-30f));
        __syncthreads();
    }
    if (lane == 0) {
        float mx = d3[0];
        for (int s = 1; s < 9; ++s) mx = fmaxf(mx, d3[s]);
        float se = 0.f, e0 = 0.f;
        for (int s = 0; s < 9; ++s) { float e = expf(d3[s] - mx); if (s == 0) e0 = e; se += e; }
        out_score[b] = (1.0f - e0 / se) * score[b];
    }
}

// bilinear 64 -> 224 (half-pixel centers, edge clamp)
__global__ __launch_bounds__(256) void upsample_kernel(const float* __restrict__ ps,
                                                       float* __restrict__ up) {
    int gid = blockIdx.x * 256 + threadIdx.x;
    if (gid >= BATCH * OUT_SZ * OUT_SZ) return;
    int b = gid / (OUT_SZ * OUT_SZ);
    int rem = gid % (OUT_SZ * OUT_SZ);
    int y = rem / OUT_SZ, x = rem % OUT_SZ;
    const float scale = (float)GRID_WH / (float)OUT_SZ;
    float sy = fminf(fmaxf(((float)y + 0.5f) * scale - 0.5f, 0.f), (float)(GRID_WH - 1));
    float sx = fminf(fmaxf(((float)x + 0.5f) * scale - 0.5f, 0.f), (float)(GRID_WH - 1));
    int y0 = (int)sy, x0 = (int)sx;
    int y1 = y0 + 1 < GRID_WH ? y0 + 1 : GRID_WH - 1;
    int x1 = x0 + 1 < GRID_WH ? x0 + 1 : GRID_WH - 1;
    float fy = sy - (float)y0, fx = sx - (float)x0;
    const float* p = ps + (size_t)b * GRID_WH * GRID_WH;
    float v00 = p[y0 * GRID_WH + x0], v01 = p[y0 * GRID_WH + x1];
    float v10 = p[y1 * GRID_WH + x0], v11 = p[y1 * GRID_WH + x1];
    float v0 = v00 + (v01 - v00) * fx, v1 = v10 + (v11 - v10) * fx;
    up[gid] = v0 + (v1 - v0) * fy;
}

// separable Gaussian sigma=4 ks=33, reflect padding
__device__ __forceinline__ int reflect_idx(int i) {
    if (i < 0) i = -i;
    if (i > OUT_SZ - 1) i = 2 * (OUT_SZ - 1) - i;
    return i;
}
template <int VERT>
__global__ __launch_bounds__(256) void blur_kernel(const float* __restrict__ in,
                                                   float* __restrict__ out) {
    __shared__ float w[33]; __shared__ float winv;
    if (threadIdx.x < 33) {
        float a = (float)threadIdx.x - 16.0f;
        w[threadIdx.x] = expf(-(a * a) / 32.0f);
    }
    __syncthreads();
    if (threadIdx.x == 0) {
        float s = 0.f;
        for (int t = 0; t < 33; ++t) s += w[t];
        winv = 1.0f / s;
    }
    __syncthreads();
    int gid = blockIdx.x * 256 + threadIdx.x;
    if (gid >= BATCH * OUT_SZ * OUT_SZ) return;
    int b = gid / (OUT_SZ * OUT_SZ);
    int rem = gid % (OUT_SZ * OUT_SZ);
    int y = rem / OUT_SZ, x = rem % OUT_SZ;
    const float* img = in + (size_t)b * OUT_SZ * OUT_SZ;
    float acc = 0.f;
#pragma unroll 11
    for (int t = 0; t < 33; ++t) {
        int yi = VERT ? reflect_idx(y + t - 16) : y;
        int xi = VERT ? x : reflect_idx(x + t - 16);
        acc += w[t] * img[yi * OUT_SZ + xi];
    }
    out[gid] = acc * winv;
}

// ---------------------------------------------------------------------------
extern "C" void kernel_launch(void* const* d_in, const int* in_sizes, int n_in,
                              void* d_out, int out_size, void* d_ws, size_t ws_size,
                              hipStream_t stream) {
    const float* E  = (const float*)d_in[0];   // [16384, 384]
    const float* Mb = (const float*)d_in[1];   // [16384, 384]
    float* out = (float*)d_out;                // [4*224*224 amap | 4 scores]

    char* w = (char*)d_ws;
    size_t off = 0;
    auto carve = [&](size_t bytes) -> void* {
        void* p = w + off;
        off = (off + bytes + 255) & ~((size_t)255);
        return p;
    };
    _Float16* Eh = (_Float16*)carve((size_t)N_ROWS * D_DIM * 2);
    _Float16* El = (_Float16*)carve((size_t)N_ROWS * D_DIM * 2);
    _Float16* Mh = (_Float16*)carve((size_t)M_ROWS * D_DIM * 2);
    _Float16* Ml = (_Float16*)carve((size_t)M_ROWS * D_DIM * 2);
    float* nE   = (float*)carve(N_ROWS * 4);
    float* nM   = (float*)carve(M_ROWS * 4);
    float* pmin = (float*)carve((size_t)N_ROWS * NSPLIT * 4);
    int*   pidx = (int*)  carve((size_t)N_ROWS * NSPLIT * 4);
    float* ps   = (float*)carve(N_ROWS * 4);
    int*   loc  = (int*)  carve(N_ROWS * 4);
    float* score  = (float*)carve(BATCH * 4);
    int*   maxidx = (int*)  carve(BATCH * 4);
    int*   nnidx  = (int*)  carve(BATCH * 4);
    float* d2   = (float*)carve((size_t)BATCH * M_ROWS * 4);
    float* up   = (float*)carve((size_t)BATCH * OUT_SZ * OUT_SZ * 4);
    float* tmp  = (float*)carve((size_t)BATCH * OUT_SZ * OUT_SZ * 4);

    const int nElem = N_ROWS * D_DIM;
    split_kernel<<<(nElem + 255) / 256, 256, 0, stream>>>(E,  Eh, El, nElem);
    split_kernel<<<(nElem + 255) / 256, 256, 0, stream>>>(Mb, Mh, Ml, nElem);
    row_norm_kernel<<<(N_ROWS * 32) / 256, 256, 0, stream>>>(E,  nE, N_ROWS);
    row_norm_kernel<<<(M_ROWS * 32) / 256, 256, 0, stream>>>(Mb, nM, M_ROWS);

    // fused WMMA distance GEMM + min/argmin: 128 row-blocks x 8 col-splits
    dist_min_kernel<<<(N_ROWS / 128) * NSPLIT, 256, 0, stream>>>(
        Eh, El, Mh, Ml, nE, nM, pmin, pidx);

    reduce_min_kernel<<<N_ROWS / 256, 256, 0, stream>>>(pmin, pidx, ps, loc);
    argmax_kernel<<<BATCH, 256, 0, stream>>>(ps, loc, score, maxidx, nnidx);
    d2_kernel<<<dim3(M_ROWS / 256, BATCH), 256, 0, stream>>>(Mb, nM, nnidx, d2);
    topk_softmax_kernel<<<BATCH, 32, 0, stream>>>(d2, E, Mb, nE, nM, score, maxidx,
                                                  out + BATCH * OUT_SZ * OUT_SZ);

    const int nPix = BATCH * OUT_SZ * OUT_SZ;
    upsample_kernel<<<(nPix + 255) / 256, 256, 0, stream>>>(ps, up);
    blur_kernel<0><<<(nPix + 255) / 256, 256, 0, stream>>>(up, tmp);   // horizontal
    blur_kernel<1><<<(nPix + 255) / 256, 256, 0, stream>>>(tmp, out);  // vertical
}
